// CosineSimCodebook_48232482734186
// MI455X (gfx1250) — matrix operand order repaired and checked by reading, not compiled, not run on hardware
//
#include <hip/hip_runtime.h>
#include <hip/hip_bf16.h>

typedef __attribute__((ext_vector_type(8)))  __bf16 v8bf;
typedef __attribute__((ext_vector_type(16))) __bf16 v16bf;
typedef __attribute__((ext_vector_type(8)))  float  v8f;

#define DECAY    0.8f
#define D        64
#define MTILE    128
#define CCHUNK   128
#define XSTRIDE  68   // padded f32 row stride (bank-conflict-free)
#define BSTRIDE  72   // padded bf16 row stride (bank-conflict-free)

static __device__ __forceinline__ v16bf join16(v8bf lo, v8bf hi) {
  return __builtin_shufflevector(lo, hi, 0,1,2,3,4,5,6,7,8,9,10,11,12,13,14,15);
}

// ---------------- Kernel 1: normalize codebook, zero accumulators ----------
__global__ void vq_prep(const float* __restrict__ embed,
                        float* __restrict__ en_f32,
                        __bf16* __restrict__ en_bf16,
                        float* __restrict__ esum,
                        float* __restrict__ bins) {
  const int lane = threadIdx.x & 31;
  const int row  = blockIdx.x * 8 + (threadIdx.x >> 5);
  const float2 v = *(const float2*)(embed + (size_t)row * D + lane * 2);
  float ss = v.x * v.x + v.y * v.y;
  #pragma unroll
  for (int m = 1; m <= 16; m <<= 1) ss += __shfl_xor(ss, m, 32);
  const float scale = 1.0f / fmaxf(sqrtf(ss), 1e-12f);
  float2 o; o.x = v.x * scale; o.y = v.y * scale;
  *(float2*)(en_f32 + (size_t)row * D + lane * 2) = o;
  en_bf16[(size_t)row * D + lane * 2]     = (__bf16)o.x;
  en_bf16[(size_t)row * D + lane * 2 + 1] = (__bf16)o.y;
  float2 z; z.x = 0.f; z.y = 0.f;
  *(float2*)(esum + (size_t)row * D + lane * 2) = z;
  if (lane == 0) bins[row] = 0.f;
}

// ---------------- Kernel 2: fused GEMM + argmax + gather + scatter ---------
__global__ __launch_bounds__(256, 2)
void vq_assign(const float*  __restrict__ x,
               const float*  __restrict__ embed_raw,
               const __bf16* __restrict__ en_bf16,
               float* __restrict__ out_q,
               float* __restrict__ out_ind,
               float* __restrict__ esum,
               float* __restrict__ bins,
               int C) {
  __shared__ float  Xn[MTILE * XSTRIDE];
  __shared__ __bf16 Xb[MTILE * BSTRIDE];
  __shared__ __bf16 Eb[CCHUNK * BSTRIDE];
  __shared__ int    BestI[MTILE];

  const int tid  = threadIdx.x;
  const int lane = tid & 31;
  const int warp = tid >> 5;
  const int m0   = blockIdx.x * MTILE;

  // Phase 1: load + l2norm 128 rows of x into LDS (2 threads / row)
  {
    const int row  = tid >> 1;
    const int part = tid & 1;
    const float* src = x + (size_t)(m0 + row) * D + part * 32;
    float v[32];
    float ss = 0.f;
    #pragma unroll
    for (int i = 0; i < 8; ++i) {
      float4 f = *(const float4*)(src + i * 4);
      v[i*4+0] = f.x; v[i*4+1] = f.y; v[i*4+2] = f.z; v[i*4+3] = f.w;
      ss += f.x*f.x + f.y*f.y + f.z*f.z + f.w*f.w;
    }
    ss += __shfl_xor(ss, 1, 32);                       // partner half-row
    const float scale = 1.0f / fmaxf(sqrtf(ss), 1e-12f);
    float*  xd = Xn + row * XSTRIDE + part * 32;
    __bf16* bd = Xb + row * BSTRIDE + part * 32;
    #pragma unroll
    for (int i = 0; i < 32; ++i) {
      const float s = v[i] * scale;
      xd[i] = s;
      bd[i] = (__bf16)s;
    }
  }
  __syncthreads();

  // A fragments (K=64 -> two 16x32 frags), resident for the whole c sweep.
  const int kh = lane >> 4;
  const __bf16* arow = Xb + (warp * 16 + (lane & 15)) * BSTRIDE;
  const v16bf a0 = join16(*(const v8bf*)(arow +      kh * 8),
                          *(const v8bf*)(arow + 16 + kh * 8));
  const v16bf a1 = join16(*(const v8bf*)(arow + 32 + kh * 8),
                          *(const v8bf*)(arow + 48 + kh * 8));

  float bestv[8];
  int   besti[8];
  #pragma unroll
  for (int r = 0; r < 8; ++r) { bestv[r] = -3.4e38f; besti[r] = 0; }

  for (int chunk = 0; chunk < C; chunk += CCHUNK) {
    __syncthreads();                                   // Eb reuse guard
    {
      const int row  = tid >> 1;
      const int part = tid & 1;
      const uint4* src = (const uint4*)(en_bf16 + (size_t)(chunk + row) * D + part * 32);
      uint4* dst = (uint4*)(Eb + row * BSTRIDE + part * 32);
      #pragma unroll
      for (int i = 0; i < 4; ++i) dst[i] = src[i];
      if (chunk + CCHUNK < C)                          // gfx1250 global_prefetch_b8
        __builtin_prefetch(en_bf16 + (size_t)(chunk + CCHUNK + row) * D + part * 32, 0, 0);
    }
    __syncthreads();

    #pragma unroll
    for (int t = 0; t < CCHUNK; t += 16) {
      const __bf16* brow = Eb + (t + (lane & 15)) * BSTRIDE;
      const v16bf b0 = join16(*(const v8bf*)(brow +      kh * 8),
                              *(const v8bf*)(brow + 16 + kh * 8));
      const v16bf b1 = join16(*(const v8bf*)(brow + 32 + kh * 8),
                              *(const v8bf*)(brow + 48 + kh * 8));
      v8f c = {};
      c = __builtin_amdgcn_wmma_f32_16x16x32_bf16(false, a0, false, b0, (short)0, c, false, false);
      c = __builtin_amdgcn_wmma_f32_16x16x32_bf16(false, a1, false, b1, (short)0, c, false, false);
      const int col = chunk + t + (lane & 15);
      #pragma unroll
      for (int r = 0; r < 8; ++r) {
        if (c[r] > bestv[r]) { bestv[r] = c[r]; besti[r] = col; }
      }
    }
  }

  // Final per-row argmax: reduce (val,idx) across the 16-lane half groups.
  #pragma unroll
  for (int r = 0; r < 8; ++r) {
    float v = bestv[r];
    int   i = besti[r];
    #pragma unroll
    for (int m = 1; m <= 8; m <<= 1) {
      const float ov = __shfl_xor(v, m, 32);
      const int   oi = __shfl_xor(i, m, 32);
      if (ov > v || (ov == v && oi < i)) { v = ov; i = oi; }
    }
    if ((lane & 15) == 0) BestI[warp * 16 + kh * 8 + r] = i;
  }
  __syncthreads();

  // quantize gather (raw codebook) — 2048 float4 per block
  for (int idx = tid; idx < MTILE * (D / 4); idx += 256) {
    const int row = idx >> 4, q = idx & 15;
    const int best = BestI[row];
    const float4 e = *((const float4*)embed_raw + (size_t)best * (D / 4) + q);
    *((float4*)out_q + (size_t)(m0 + row) * (D / 4) + q) = e;
  }
  // indices (as float) + bins histogram
  if (tid < MTILE) {
    const int best = BestI[tid];
    out_ind[m0 + tid] = (float)best;
    atomicAdd(bins + best, 1.0f);
  }
  // embed_sum scatter of normalized rows
  for (int idx = tid; idx < MTILE * D; idx += 256) {
    const int row = idx >> 6, k = idx & 63;
    atomicAdd(esum + (size_t)BestI[row] * D + k, Xn[row * XSTRIDE + k]);
  }
}

// ---------------- Kernel 3: EMA update of codebook + cluster size ----------
__global__ void vq_update(const float* __restrict__ embed,
                          const float* __restrict__ csize,
                          const float* __restrict__ en_f32,
                          const float* __restrict__ esum,
                          const float* __restrict__ bins,
                          float* __restrict__ out_embed,
                          float* __restrict__ out_cs) {
  const int lane = threadIdx.x & 31;
  const int c    = blockIdx.x * 8 + (threadIdx.x >> 5);
  const float b = bins[c];
  if (lane == 0) out_cs[c] = csize[c] * DECAY + b * (1.0f - DECAY);
  const float bsafe = (b == 0.f) ? 1.f : b;
  float2 s = *(const float2*)(esum + (size_t)c * D + lane * 2);
  s.x /= bsafe; s.y /= bsafe;
  float ss = s.x * s.x + s.y * s.y;
  #pragma unroll
  for (int m = 1; m <= 16; m <<= 1) ss += __shfl_xor(ss, m, 32);
  const float scale = 1.0f / fmaxf(sqrtf(ss), 1e-12f);
  const float2 en = *(const float2*)(en_f32 + (size_t)c * D + lane * 2);
  const float sx = (b == 0.f) ? en.x : s.x * scale;
  const float sy = (b == 0.f) ? en.y : s.y * scale;
  const float2 e = *(const float2*)(embed + (size_t)c * D + lane * 2);
  float2 o;
  o.x = e.x * DECAY + sx * (1.f - DECAY);
  o.y = e.y * DECAY + sy * (1.f - DECAY);
  *(float2*)(out_embed + (size_t)c * D + lane * 2) = o;
}

// ---------------- launch --------------------------------------------------
extern "C" void kernel_launch(void* const* d_in, const int* in_sizes, int n_in,
                              void* d_out, int out_size, void* d_ws, size_t ws_size,
                              hipStream_t stream) {
  const float* x     = (const float*)d_in[0];   // [16,4096,64]
  const float* embed = (const float*)d_in[1];   // [1,4096,64]
  const float* csize = (const float*)d_in[2];   // [1,4096]
  const int N = in_sizes[0] / D;                // 65536
  const int C = in_sizes[2];                    // 4096

  float* out       = (float*)d_out;
  float* out_q     = out;                                   // [16,4096,64]
  float* out_ind   = out + (size_t)N * D;                   // [16,4096] (as float)
  float* out_embed = out_ind + N;                           // [1,4096,64]
  float* out_cs    = out_embed + (size_t)C * D;             // [1,4096]

  float*  en_f32  = (float*)d_ws;
  float*  esum    = en_f32 + (size_t)C * D;
  float*  bins    = esum + (size_t)C * D;
  __bf16* en_bf16 = (__bf16*)(bins + C);                    // 16B-aligned

  vq_prep  <<<C / 8,     256, 0, stream>>>(embed, en_f32, en_bf16, esum, bins);
  vq_assign<<<N / MTILE, 256, 0, stream>>>(x, embed, en_bf16, out_q, out_ind, esum, bins, C);
  vq_update<<<C / 8,     256, 0, stream>>>(embed, csize, en_f32, esum, bins, out_embed, out_cs);
}
